// VQuantizer_14877766713515
// MI455X (gfx1250) — matrix operand order repaired and checked by reading, not compiled, not run on hardware
//
#include <hip/hip_runtime.h>
#include <math.h>

// ---------------------------------------------------------------------------
// VQ-transformer forward for MI455X (gfx1250, wave32, WMMA f16 16x16x32).
// f16 activation pipeline: GEMM/attention operands are f16 (pure b128 vector
// loads + v_wmma in the inner loops); fp32 residual stream with f16 shadow.
// V is pre-transposed to [nb][h][dh][S] so the PV B-fragments are contiguous
// 32-byte loads, same as the GEMM B path.
// ---------------------------------------------------------------------------

typedef __attribute__((ext_vector_type(16))) _Float16 v16h;
typedef __attribute__((ext_vector_type(8)))  _Float16 v8h;
typedef __attribute__((ext_vector_type(8)))  float    v8f;

// CDNA5 WMMA fragment index maps (ISA 7.12.2):
// A 16x32 f16: lane<16 -> m=lane, K groups {0..7,16..23}; lane>=16 -> {8..15,24..31}
__device__ __forceinline__ int a_k(int lane, int e) {
    return ((lane >> 4) << 3) + (e & 7) + ((e & 8) << 1);
}
// B 32x16 f16, contiguous k order used by this code (A must match): n = lane&15; lanes 0-15 hold K=0..15, lanes 16-31 hold K=16..31
__device__ __forceinline__ int b_k(int lane, int e) {
    return ((lane >> 4) << 4) + e;
}
// C/D 16x16 f32: element r -> m = r + 8*(lane>=16), n = lane&15

// A fragment from a contiguous f16 row (kb = 8*(lane>=16)): groups
// [kb, kb+8) and [kb+16, kb+24) -> two aligned 16B loads, no conversion.
__device__ __forceinline__ v16h load_a_h(const _Float16* __restrict__ row, int kb) {
    const v8h lo = *(const v8h*)(row + kb);
    const v8h hi = *(const v8h*)(row + kb + 16);
    v16h a;
#pragma unroll
    for (int e = 0; e < 8; ++e) { a[e] = lo[e]; a[e + 8] = hi[e]; }
    return a;
}

// ---------------------------------------------------------------------------
// Wave-tiled GEMM, 16x64 per wave: out = A @ W^T (+bias) (+relu)
//   A: M x Kd row-major f16 (row = s*NB+nb)
//   W: N x Kd row-major f16 (transposed-weight layout)
//   out (fp32 or f16) element (m, n): out[s*so_s + nb*so_n + col_off + n]
// ---------------------------------------------------------------------------
template <typename TO, bool RELU>
__global__ __launch_bounds__(256) void gemm_kernel(
    const _Float16* __restrict__ A, int lda,
    const _Float16* __restrict__ W, int ldw,
    const float* __restrict__ bias,
    TO* __restrict__ out,
    int M, int N, int Kd,
    long so_s, long so_n, int col_off, int NBv)
{
    const int lane = threadIdx.x & 31;
    const int wv   = blockIdx.x * (blockDim.x >> 5) + (threadIdx.x >> 5);
    const int nT   = N >> 6;                 // 64-wide n blocks
    const int total = (M >> 4) * nT;
    if (wv >= total) return;
    const int mt = wv / nT;
    const int nt = wv - mt * nT;

    const _Float16* Arow = A + (long)(mt * 16 + (lane & 15)) * lda;
    const _Float16* Wr[4];
#pragma unroll
    for (int j = 0; j < 4; ++j)
        Wr[j] = W + (long)(nt * 64 + j * 16 + (lane & 15)) * ldw;

    const int kb  = ((lane >> 4) << 3);
    const int b16 = ((lane >> 4) << 4);

    v8f acc[4] = {};
    for (int k0 = 0; k0 < Kd; k0 += 32) {
        __builtin_prefetch(Arow + k0 + 256, 0, 0);   // global_prefetch_b8
        const v16h a = *(const v16h*)(Arow + k0 + b16);
#pragma unroll
        for (int j = 0; j < 4; ++j) {
            const v16h b = *(const v16h*)(Wr[j] + k0 + b16);   // contiguous 32B
            acc[j] = __builtin_amdgcn_wmma_f32_16x16x32_f16(
                false, a, false, b, (short)0, acc[j], false, false);
        }
    }

#pragma unroll
    for (int j = 0; j < 4; ++j) {
        const int n  = nt * 64 + j * 16 + (lane & 15);
        const float bv = bias ? bias[n] : 0.0f;
#pragma unroll
        for (int r = 0; r < 8; ++r) {
            const int m  = mt * 16 + r + ((lane >> 4) << 3);
            const int s  = m / NBv;
            const int nb = m - s * NBv;
            float v = acc[j][r] + bv;
            if (RELU) v = fmaxf(v, 0.0f);
            out[(long)s * so_s + (long)nb * so_n + col_off + n] = (TO)v;
        }
    }
}

// V transpose: qkv V-section (rows (t*NB+nb) x 3C) -> vt[nb][h][d][t]
__global__ __launch_bounds__(256) void vtrans_kernel(
    const _Float16* __restrict__ qkv, _Float16* __restrict__ vt,
    long total, int S, int NBv, int C, int NHv)
{
    constexpr int DH = 64;
    const long idx = (long)blockIdx.x * 256 + threadIdx.x;
    if (idx >= total) return;
    const int t = (int)(idx % S);
    long u = idx / S;
    const int d = (int)(u % DH); u /= DH;
    const int h = (int)(u % NHv); u /= NHv;
    const int nb = (int)u;
    vt[idx] = qkv[((long)t * NBv + nb) * 3 * C + 2 * C + h * DH + d];
}

// ---------------------------------------------------------------------------
// Flash-style attention on f16 qkv (Q/K) + transposed V (vt [nb][h][dh][S]).
// One wave per (nb, h, 16-query block); keys/values in chunks of 32.
// Output written as f16 (consumed only by the out-projection GEMM).
// ---------------------------------------------------------------------------
__global__ __launch_bounds__(256) void attn_kernel(
    const _Float16* __restrict__ qkv, const _Float16* __restrict__ vt,
    _Float16* __restrict__ out, int S, int NBv, int C, int NHv)
{
    constexpr int DH = 64;
    __shared__ _Float16 pbuf[8][16][34];   // per-wave 16x32 P tile, padded

    const int lane = threadIdx.x & 31;
    const int w    = threadIdx.x >> 5;
    const int gw   = blockIdx.x * (blockDim.x >> 5) + w;
    const int total = (S >> 4) * NBv * NHv;
    if (gw >= total) return;

    const int h  = gw % NHv;
    const int nb = (gw / NHv) % NBv;
    const int qt = gw / (NHv * NBv);

    const long NB3C = (long)NBv * 3 * C;
    const _Float16* qp = qkv + (long)nb * 3 * C + h * DH;
    const _Float16* kp = qp + C;
    const int kb  = ((lane >> 4) << 3);
    const int b16 = ((lane >> 4) << 4);

    // V row pointers (transposed layout): row = d-column, contiguous in t
    const _Float16* vr[4];
#pragma unroll
    for (int d = 0; d < 4; ++d)
        vr[d] = vt + (((long)nb * NHv + h) * DH + d * 16 + (lane & 15)) * S;

    // Preload two Q A-fragments (k=0..31, 32..63), scaled by 1/sqrt(dh)=0.125
    const _Float16 qs = (_Float16)0.125f;
    const _Float16* qrow = qp + (long)(qt * 16 + (lane & 15)) * NB3C;
    const v8h q0a = *(const v8h*)(qrow + kb);
    const v8h q0b = *(const v8h*)(qrow + kb + 16);
    const v8h q1a = *(const v8h*)(qrow + kb + 32);
    const v8h q1b = *(const v8h*)(qrow + kb + 48);
    v16h aq0, aq1;
#pragma unroll
    for (int e = 0; e < 8; ++e) {
        aq0[e] = q0a[e] * qs;  aq0[e + 8] = q0b[e] * qs;
        aq1[e] = q1a[e] * qs;  aq1[e + 8] = q1b[e] * qs;
    }

    float rmax[8], rsum[8];
#pragma unroll
    for (int r = 0; r < 8; ++r) { rmax[r] = -__builtin_inff(); rsum[r] = 0.0f; }
    v8f o0 = {}, o1 = {}, o2 = {}, o3 = {};

    for (int t0 = 0; t0 < S; t0 += 32) {
        // scores: two 16x16 tiles (t cols t0..t0+15, t0+16..t0+31)
        v8f sc[2];
#pragma unroll
        for (int tt = 0; tt < 2; ++tt) {
            const _Float16* krow = kp + (long)(t0 + tt * 16 + (lane & 15)) * NB3C;
            const v8h klo0 = *(const v8h*)(krow + kb);
            const v8h khi0 = *(const v8h*)(krow + kb + 16);
            const v8h klo1 = *(const v8h*)(krow + kb + 32);
            const v8h khi1 = *(const v8h*)(krow + kb + 48);
            v16h b0, b1;
#pragma unroll
            for (int e = 0; e < 8; ++e) {
                b0[e] = klo0[e]; b0[e + 8] = khi0[e];
                b1[e] = klo1[e]; b1[e + 8] = khi1[e];
            }
            v8f z = {};
            z = __builtin_amdgcn_wmma_f32_16x16x32_f16(false, aq0, false, b0, (short)0, z, false, false);
            z = __builtin_amdgcn_wmma_f32_16x16x32_f16(false, aq1, false, b1, (short)0, z, false, false);
            sc[tt] = z;
        }

        // online softmax: row max across the 16-lane half (n = lane&15)
        float nm[8];
#pragma unroll
        for (int r = 0; r < 8; ++r) nm[r] = fmaxf(sc[0][r], sc[1][r]);
#pragma unroll
        for (int off = 1; off < 16; off <<= 1)
#pragma unroll
            for (int r = 0; r < 8; ++r) nm[r] = fmaxf(nm[r], __shfl_xor(nm[r], off, 32));

        const int prow = ((lane >> 4) << 3);
#pragma unroll
        for (int r = 0; r < 8; ++r) {
            const float m2   = fmaxf(rmax[r], nm[r]);
            const float corr = __expf(rmax[r] - m2);
            rmax[r] = m2;
            rsum[r] *= corr;
            o0[r] *= corr; o1[r] *= corr; o2[r] *= corr; o3[r] *= corr;
            const float p0 = __expf(sc[0][r] - m2);
            const float p1 = __expf(sc[1][r] - m2);
            rsum[r] += p0 + p1;
            pbuf[w][r + prow][(lane & 15)]      = (_Float16)p0;
            pbuf[w][r + prow][(lane & 15) + 16] = (_Float16)p1;
        }
        asm volatile("s_wait_dscnt 0" ::: "memory");

        // reload P as A fragment (16x32, K = t within chunk)
        v16h pa;
#pragma unroll
        for (int e = 0; e < 16; ++e) pa[e] = pbuf[w][lane & 15][b_k(lane, e)];

        // PV: 4 output tiles across dh=64; B fragments are contiguous 32B
#pragma unroll
        for (int d = 0; d < 4; ++d) {
            const v16h bv = *(const v16h*)(vr[d] + t0 + b16);
            v8f* od = (d == 0) ? &o0 : (d == 1) ? &o1 : (d == 2) ? &o2 : &o3;
            *od = __builtin_amdgcn_wmma_f32_16x16x32_f16(false, pa, false, bv, (short)0, *od, false, false);
        }
    }

    // finalize: reduce row sums across the 16-lane half, normalize, store f16
#pragma unroll
    for (int r = 0; r < 8; ++r)
#pragma unroll
        for (int off = 1; off < 16; off <<= 1) rsum[r] += __shfl_xor(rsum[r], off, 32);

#pragma unroll
    for (int r = 0; r < 8; ++r) {
        const float inv = 1.0f / rsum[r];
        const long  m    = qt * 16 + r + ((lane >> 4) << 3);
        const long  base = (m * NBv + nb) * (long)C + h * DH;
        out[base +  0 + (lane & 15)] = (_Float16)(o0[r] * inv);
        out[base + 16 + (lane & 15)] = (_Float16)(o1[r] * inv);
        out[base + 32 + (lane & 15)] = (_Float16)(o2[r] * inv);
        out[base + 48 + (lane & 15)] = (_Float16)(o3[r] * inv);
    }
}

// ---------------------------------------------------------------------------
// LayerNorm (fused residual): out = LN(a + b) * g + beta. Wave per row.
// Writes fp32 residual stream and f16 shadow copy for the GEMM/attention path.
// ---------------------------------------------------------------------------
__global__ __launch_bounds__(256) void ln_kernel(
    const float* __restrict__ a, const float* __restrict__ b,
    const float* __restrict__ g, const float* __restrict__ be,
    float* __restrict__ out, _Float16* __restrict__ out_h, int rows, int C)
{
    const int lane = threadIdx.x & 31;
    const int row  = blockIdx.x * (blockDim.x >> 5) + (threadIdx.x >> 5);
    if (row >= rows) return;
    const float* ar = a + (long)row * C;
    const float* br = b ? b + (long)row * C : nullptr;
    float v[16];
    float s = 0.0f;
#pragma unroll
    for (int i = 0; i < 16; ++i) {
        float t = ar[lane + i * 32];
        if (br) t += br[lane + i * 32];
        v[i] = t; s += t;
    }
#pragma unroll
    for (int off = 1; off < 32; off <<= 1) s += __shfl_xor(s, off, 32);
    const float mean = s / (float)C;
    float vs = 0.0f;
#pragma unroll
    for (int i = 0; i < 16; ++i) { const float d = v[i] - mean; vs += d * d; }
#pragma unroll
    for (int off = 1; off < 32; off <<= 1) vs += __shfl_xor(vs, off, 32);
    const float rstd = rsqrtf(vs / (float)C + 1e-5f);
    float* orow = out + (long)row * C;
    _Float16* hrow = out_h + (long)row * C;
#pragma unroll
    for (int i = 0; i < 16; ++i) {
        const int c = lane + i * 32;
        const float y = (v[i] - mean) * rstd * g[c] + be[c];
        orow[c] = y;
        hrow[c] = (_Float16)y;
    }
}

// ---------------------------------------------------------------------------
// Row softmax over K logits (output layout (nb,s,k)): f16 probs, argmax code
// (first-index tie-break), codes as float into d_out. One block per row.
// ---------------------------------------------------------------------------
__global__ __launch_bounds__(256) void rowstat_kernel(
    const float* __restrict__ logits, _Float16* __restrict__ probs,
    int* __restrict__ codes_i, float* __restrict__ codes_f,
    int S, int NBv, int K)
{
    __shared__ float sv[256];
    __shared__ int   si[256];
    __shared__ float red;
    const int r  = blockIdx.x;
    const int s  = r / NBv;
    const int nb = r - s * NBv;
    const float* row = logits + (long)nb * S * K + (long)s * K;
    const int t = threadIdx.x;

    float bm = -__builtin_inff(); int bi = 0;
    for (int k = t; k < K; k += 256) {
        const float v = row[k];
        if (v > bm) { bm = v; bi = k; }
    }
    sv[t] = bm; si[t] = bi;
    __syncthreads();
    for (int off = 128; off > 0; off >>= 1) {
        if (t < off) {
            const float v2 = sv[t + off]; const int i2 = si[t + off];
            if (v2 > sv[t] || (v2 == sv[t] && i2 < si[t])) { sv[t] = v2; si[t] = i2; }
        }
        __syncthreads();
    }
    if (t == 0) {
        red = sv[0];
        codes_i[r] = si[0];
        codes_f[(long)nb * S + s] = (float)si[0];
    }
    __syncthreads();
    const float mx = red;

    float ls = 0.0f;
    for (int k = t; k < K; k += 256) ls += __expf(row[k] - mx);
    sv[t] = ls;
    __syncthreads();
    for (int off = 128; off > 0; off >>= 1) {
        if (t < off) sv[t] += sv[t + off];
        __syncthreads();
    }
    const float inv = 1.0f / sv[0];
    _Float16* pr = probs + (long)r * K;
    for (int k = t; k < K; k += 256) pr[k] = (_Float16)(__expf(row[k] - mx) * inv);
}

// hard = cb_w.T[code]; write f16 activation (for post GEMM) + (n,c,h,w) output
__global__ __launch_bounds__(256) void hard_kernel(
    const int* __restrict__ codes_i, const float* __restrict__ cb,
    _Float16* __restrict__ hard_h, float* __restrict__ out_hards,
    long total, int C, int K, int NBv, int S)
{
    const long idx = (long)blockIdx.x * 256 + threadIdx.x;
    if (idx >= total) return;
    const int c = (int)(idx % C);
    const int r = (int)(idx / C);
    const int code = codes_i[r];
    const float v = cb[(long)c * K + code];
    hard_h[idx] = (_Float16)v;
    const int s = r / NBv, nb = r - (r / NBv) * NBv;
    out_hards[((long)nb * C + c) * S + s] = v;
}

// (S,n,C) row-major -> (n,C,S)
__global__ __launch_bounds__(256) void tout_kernel(
    const float* __restrict__ act, float* __restrict__ out,
    long total, int C, int NBv, int S)
{
    const long idx = (long)blockIdx.x * 256 + threadIdx.x;
    if (idx >= total) return;
    const int c = (int)(idx % C);
    const int r = (int)(idx / C);
    const int s = r / NBv, nb = r - s * NBv;
    out[((long)nb * C + c) * S + s] = act[idx];
}

// x (n,c,h,w) -> enc_flat (S,n,c) and posisted (f32 + f16) with 2D sinusoidal PE
__global__ __launch_bounds__(256) void pe_kernel(
    const float* __restrict__ x, float* __restrict__ enc_flat,
    float* __restrict__ act, _Float16* __restrict__ act_h,
    long total, int NBv, int C, int H, int Wd)
{
    const long idx = (long)blockIdx.x * 256 + threadIdx.x;
    if (idx >= total) return;
    const int  c   = (int)(idx % C);
    const long t   = idx / C;
    const int  nb  = (int)(t % NBv);
    const int  sp  = (int)(t / NBv);
    const int  hh  = sp / Wd, ww = sp - hh * Wd;
    const float v = x[(((long)nb * C + c) * H + hh) * Wd + ww];
    enc_flat[idx] = v;
    const int half = C >> 1;
    int pos, cc;
    if (c < half) { pos = ww; cc = c; } else { pos = hh; cc = c - half; }
    const int i2 = (cc >> 1) << 1;
    const float div = __expf(-(float)i2 * __logf(10000.0f) / (float)half);
    const float ang = (float)pos * div;
    const float pe = (cc & 1) ? __cosf(ang) : __sinf(ang);
    const float y = v + pe;
    act[idx] = y;
    act_h[idx] = (_Float16)y;
}

// decoder input: mask ? enc_flat : post  (mask per (s,nb) row; jax bool = u8)
__global__ __launch_bounds__(256) void mix_kernel(
    const unsigned char* __restrict__ mask, const float* __restrict__ enc_flat,
    const float* __restrict__ mem, float* __restrict__ dst,
    _Float16* __restrict__ dst_h, long total, int C)
{
    const long idx = (long)blockIdx.x * 256 + threadIdx.x;
    if (idx >= total) return;
    const int r = (int)(idx / C);
    const float v = mask[r] ? enc_flat[idx] : mem[idx];
    dst[idx] = v;
    dst_h[idx] = (_Float16)v;
}

__global__ __launch_bounds__(256) void cvt_kernel(
    const float* __restrict__ src, _Float16* __restrict__ dst, long n)
{
    const long idx = (long)blockIdx.x * 256 + threadIdx.x;
    if (idx < n) dst[idx] = (_Float16)src[idx];
}

// transpose-convert: src (R x Kc) f32 -> dst (Kc x R) f16
__global__ __launch_bounds__(256) void tcvt_kernel(
    const float* __restrict__ src, _Float16* __restrict__ dst, int R, int Kc)
{
    const long idx = (long)blockIdx.x * 256 + threadIdx.x;
    if (idx >= (long)R * Kc) return;
    const int k = (int)(idx % Kc);
    const int c = (int)(idx / Kc);
    dst[(long)k * R + c] = (_Float16)src[idx];
}

// ---------------------------------------------------------------------------
extern "C" void kernel_launch(void* const* d_in, const int* in_sizes, int n_in,
                              void* d_out, int out_size, void* d_ws, size_t ws_size,
                              hipStream_t stream)
{
    (void)in_sizes; (void)n_in; (void)out_size; (void)ws_size;
    constexpr int NBv = 8, C = 512, H = 32, Wd = 32, K = 4096, L = 6, NHv = 8;
    constexpr int S = H * Wd, ROWS = S * NBv, C3 = 3 * C;
    constexpr long AELT = (long)ROWS * C;

    // inputs (setup_inputs order)
    const float* x        = (const float*)d_in[0];
    const float* pre_w    = (const float*)d_in[1];
    const float* pre_b    = (const float*)d_in[2];
    const float* post_w   = (const float*)d_in[3];
    const float* post_b   = (const float*)d_in[4];
    const float* cb_w     = (const float*)d_in[5];
    const float* e_qkv_w  = (const float*)d_in[6];
    const float* e_qkv_b  = (const float*)d_in[7];
    const float* e_o_w    = (const float*)d_in[8];
    const float* e_o_b    = (const float*)d_in[9];
    const float* e_ln1_w  = (const float*)d_in[10];
    const float* e_ln1_b  = (const float*)d_in[11];
    const float* e_ln2_w  = (const float*)d_in[12];
    const float* e_ln2_b  = (const float*)d_in[13];
    const float* e_f1_w   = (const float*)d_in[14];
    const float* e_f1_b   = (const float*)d_in[15];
    const float* e_f2_w   = (const float*)d_in[16];
    const float* e_f2_b   = (const float*)d_in[17];
    const float* d_sqkv_w = (const float*)d_in[18];
    const float* d_sqkv_b = (const float*)d_in[19];
    const float* d_so_w   = (const float*)d_in[20];
    const float* d_so_b   = (const float*)d_in[21];
    const float* d_cqkv_w = (const float*)d_in[22];
    const float* d_cqkv_b = (const float*)d_in[23];
    const float* d_co_w   = (const float*)d_in[24];
    const float* d_co_b   = (const float*)d_in[25];
    const float* d_ln1_w  = (const float*)d_in[26];
    const float* d_ln1_b  = (const float*)d_in[27];
    const float* d_ln2_w  = (const float*)d_in[28];
    const float* d_ln2_b  = (const float*)d_in[29];
    const float* d_ln3_w  = (const float*)d_in[30];
    const float* d_ln3_b  = (const float*)d_in[31];
    const float* d_f1_w   = (const float*)d_in[32];
    const float* d_f1_b   = (const float*)d_in[33];
    const float* d_f2_w   = (const float*)d_in[34];
    const float* d_f2_b   = (const float*)d_in[35];
    const unsigned char* mask = (const unsigned char*)d_in[36];

    // output regions (return order)
    float* out_quant  = (float*)d_out;
    float* out_high   = out_quant + AELT;
    float* out_softs  = out_high  + AELT;
    float* out_hards  = out_softs + AELT;
    float* out_codes  = out_hards + AELT;          // ROWS elements
    float* out_logits = out_codes + ROWS;          // ROWS*K elements

    // workspace carve-up
    char* wp = (char*)d_ws;
    auto ah = [&](size_t n) { _Float16* p = (_Float16*)wp; wp += n * sizeof(_Float16); return p; };
    auto af = [&](size_t n) { float* p = (float*)wp; wp += n * sizeof(float); return p; };

    _Float16* preW  = ah((size_t)C * C);
    _Float16* postW = ah((size_t)C * C);
    _Float16* cbW   = ah((size_t)C * K);   // (C,K): soft GEMM B (N=C rows, ld=K)
    _Float16* cbT   = ah((size_t)K * C);   // (K,C): sim GEMM B (N=K rows, ld=C)
    _Float16* eQkvW = ah((size_t)L * C3 * C);
    _Float16* eOW   = ah((size_t)L * C * C);
    _Float16* eF1W  = ah((size_t)L * C * C);
    _Float16* eF2W  = ah((size_t)L * C * C);
    _Float16* dSqkvW= ah((size_t)L * C3 * C);
    _Float16* dSoW  = ah((size_t)L * C * C);
    _Float16* dCqkvW= ah((size_t)L * C3 * C);
    _Float16* dCoW  = ah((size_t)L * C * C);
    _Float16* dF1W  = ah((size_t)L * C * C);
    _Float16* dF2W  = ah((size_t)L * C * C);
    _Float16* probs = ah((size_t)ROWS * K);
    _Float16* qkv_h   = ah((size_t)ROWS * C3);
    _Float16* vt_h    = ah(AELT);
    _Float16* attnb_h = ah(AELT);
    _Float16* hid_h   = ah(AELT);
    _Float16* act_h   = ah(AELT);
    _Float16* mem_h   = ah(AELT);
    _Float16* pre_h   = ah(AELT);
    _Float16* hard_h  = ah(AELT);
    float* enc_flat = af(AELT);
    float* act      = af(AELT);
    float* tmp      = af(AELT);
    float* mem      = af(AELT);
    float* pre_act  = af(AELT);
    int*   codes_i  = (int*)af(ROWS);

    auto cvt = [&](const float* src, _Float16* dst, size_t n) {
        cvt_kernel<<<(int)((n + 255) / 256), 256, 0, stream>>>(src, dst, (long)n);
    };
    cvt(pre_w, preW, (size_t)C * C);
    cvt(post_w, postW, (size_t)C * C);
    cvt(cb_w, cbW, (size_t)C * K);
    tcvt_kernel<<<(int)(((size_t)C * K + 255) / 256), 256, 0, stream>>>(cb_w, cbT, C, K);
    cvt(e_qkv_w, eQkvW, (size_t)L * C3 * C);
    cvt(e_o_w,  eOW,  (size_t)L * C * C);
    cvt(e_f1_w, eF1W, (size_t)L * C * C);
    cvt(e_f2_w, eF2W, (size_t)L * C * C);
    cvt(d_sqkv_w, dSqkvW, (size_t)L * C3 * C);
    cvt(d_so_w,  dSoW,  (size_t)L * C * C);
    cvt(d_cqkv_w, dCqkvW, (size_t)L * C3 * C);
    cvt(d_co_w,  dCoW,  (size_t)L * C * C);
    cvt(d_f1_w, dF1W, (size_t)L * C * C);
    cvt(d_f2_w, dF2W, (size_t)L * C * C);

    // GEMM launchers: A f16, W f16; out fp32 or f16, row-major with stride ldo
    auto gemmF = [&](const _Float16* A, int lda, const _Float16* Wm, int ldw,
                     const float* bias, float* out, int M, int N, int Kd,
                     int ldo, int coff) {
        const int blocks = ((M / 16) * (N / 64) + 7) / 8;
        gemm_kernel<float, false><<<blocks, 256, 0, stream>>>(
            A, lda, Wm, ldw, bias, out, M, N, Kd,
            (long)NBv * ldo, (long)ldo, coff, NBv);
    };
    auto gemmH = [&](const _Float16* A, int lda, const _Float16* Wm, int ldw,
                     const float* bias, _Float16* out, int M, int N, int Kd,
                     int ldo, int coff, bool relu) {
        const int blocks = ((M / 16) * (N / 64) + 7) / 8;
        if (relu)
            gemm_kernel<_Float16, true><<<blocks, 256, 0, stream>>>(
                A, lda, Wm, ldw, bias, out, M, N, Kd,
                (long)NBv * ldo, (long)ldo, coff, NBv);
        else
            gemm_kernel<_Float16, false><<<blocks, 256, 0, stream>>>(
                A, lda, Wm, ldw, bias, out, M, N, Kd,
                (long)NBv * ldo, (long)ldo, coff, NBv);
    };
    const int eltBlocks = (int)((AELT + 255) / 256);
    auto attn = [&](const _Float16* qkvB, _Float16* ob) {
        vtrans_kernel<<<eltBlocks, 256, 0, stream>>>(qkvB, vt_h, AELT, S, NBv, C, NHv);
        const int totalW = (S / 16) * NBv * NHv;
        attn_kernel<<<(totalW + 7) / 8, 256, 0, stream>>>(qkvB, vt_h, ob, S, NBv, C, NHv);
    };
    auto ln = [&](const float* a, const float* b, const float* g, const float* be,
                  float* o, _Float16* oh) {
        ln_kernel<<<(ROWS + 7) / 8, 256, 0, stream>>>(a, b, g, be, o, oh, ROWS, C);
    };

    // ---- encoder input: transpose + positional encoding ----
    pe_kernel<<<eltBlocks, 256, 0, stream>>>(x, enc_flat, act, act_h, AELT, NBv, C, H, Wd);

    // ---- encoder ----
    for (int i = 0; i < L; ++i) {
        gemmH(act_h, C, eQkvW + (size_t)i * C3 * C, C, e_qkv_b + (size_t)i * C3,
              qkv_h, ROWS, C3, C, C3, 0, false);
        attn(qkv_h, attnb_h);
        gemmF(attnb_h, C, eOW + (size_t)i * C * C, C, e_o_b + (size_t)i * C,
              tmp, ROWS, C, C, C, 0);
        ln(act, tmp, e_ln1_w + (size_t)i * C, e_ln1_b + (size_t)i * C, act, act_h);
        gemmH(act_h, C, eF1W + (size_t)i * C * C, C, e_f1_b + (size_t)i * C,
              hid_h, ROWS, C, C, C, 0, true);
        gemmF(hid_h, C, eF2W + (size_t)i * C * C, C, e_f2_b + (size_t)i * C,
              tmp, ROWS, C, C, C, 0);
        ln(act, tmp, e_ln2_w + (size_t)i * C, e_ln2_b + (size_t)i * C, act, act_h);
    }

    // ---- preProcess, codebook ----
    gemmF(act_h, C, preW, C, pre_b, pre_act, ROWS, C, C, C, 0);
    cvt(pre_act, pre_h, (size_t)AELT);
    tout_kernel<<<eltBlocks, 256, 0, stream>>>(pre_act, out_high, AELT, C, NBv, S);

    // sim -> logits in (n, s, k) order; B = cbT (K x C, transposed layout)
    {
        const int blocks = ((ROWS / 16) * (K / 64) + 7) / 8;
        gemm_kernel<float, false><<<blocks, 256, 0, stream>>>(
            pre_h, C, cbT, C, (const float*)nullptr, out_logits,
            ROWS, K, C, (long)K, (long)S * K, 0, NBv);
    }
    rowstat_kernel<<<ROWS, 256, 0, stream>>>(out_logits, probs, codes_i, out_codes, S, NBv, K);
    hard_kernel<<<eltBlocks, 256, 0, stream>>>(codes_i, cb_w, hard_h, out_hards, AELT, C, K, NBv, S);

    // soft = probs @ cb_w^T (B = cbW (C x K)), then transpose out
    gemmF(probs, K, cbW, K, (const float*)nullptr, tmp, ROWS, C, K, C, 0);
    tout_kernel<<<eltBlocks, 256, 0, stream>>>(tmp, out_softs, AELT, C, NBv, S);

    // quant == hard numerically; postProcess -> mem (f32 for mix, f16 for GEMM)
    gemmF(hard_h, C, postW, C, post_b, mem, ROWS, C, C, C, 0);
    cvt(mem, mem_h, (size_t)AELT);
    mix_kernel<<<eltBlocks, 256, 0, stream>>>(mask, enc_flat, mem, act, act_h, AELT, C);

    // ---- decoder ----
    for (int i = 0; i < L; ++i) {
        // self attention
        gemmH(act_h, C, dSqkvW + (size_t)i * C3 * C, C, d_sqkv_b + (size_t)i * C3,
              qkv_h, ROWS, C3, C, C3, 0, false);
        attn(qkv_h, attnb_h);
        gemmF(attnb_h, C, dSoW + (size_t)i * C * C, C, d_so_b + (size_t)i * C,
              tmp, ROWS, C, C, C, 0);
        ln(act, tmp, d_ln1_w + (size_t)i * C, d_ln1_b + (size_t)i * C, act, act_h);
        // cross attention: q from act, k/v from mem (shared packed weight)
        gemmH(act_h, C, dCqkvW + (size_t)i * C3 * C, C, d_cqkv_b + (size_t)i * C3,
              qkv_h, ROWS, C, C, C3, 0, false);
        gemmH(mem_h, C, dCqkvW + (size_t)i * C3 * C + (size_t)C * C, C,
              d_cqkv_b + (size_t)i * C3 + C, qkv_h, ROWS, 2 * C, C, C3, C, false);
        attn(qkv_h, attnb_h);
        gemmF(attnb_h, C, dCoW + (size_t)i * C * C, C, d_co_b + (size_t)i * C,
              tmp, ROWS, C, C, C, 0);
        ln(act, tmp, d_ln2_w + (size_t)i * C, d_ln2_b + (size_t)i * C, act, act_h);
        // FFN
        gemmH(act_h, C, dF1W + (size_t)i * C * C, C, d_f1_b + (size_t)i * C,
              hid_h, ROWS, C, C, C, 0, true);
        gemmF(hid_h, C, dF2W + (size_t)i * C * C, C, d_f2_b + (size_t)i * C,
              tmp, ROWS, C, C, C, 0);
        ln(act, tmp, d_ln3_w + (size_t)i * C, d_ln3_b + (size_t)i * C, act, act_h);
    }

    tout_kernel<<<eltBlocks, 256, 0, stream>>>(act, out_quant, AELT, C, NBv, S);
}
